// MAB_53764400611374
// MI455X (gfx1250) — matrix operand order repaired and checked by reading, not compiled, not run on hardware
//
#include <hip/hip_runtime.h>
#include <math.h>

typedef __bf16 v16bf __attribute__((ext_vector_type(16)));
typedef float  v8f   __attribute__((ext_vector_type(8)));

#define B_     8
#define C_     256
#define HW_    1024
#define NPIX   (B_ * HW_)        // 8192
#define NH_    8
#define HD_    32
#define HID_   512

// ---------------------------------------------------------------------------
// f32 -> bf16 weight cast
// ---------------------------------------------------------------------------
__global__ void k_cast_bf16(const float* __restrict__ in, __bf16* __restrict__ out, int n) {
    int i = blockIdx.x * 256 + threadIdx.x;
    if (i < n) out[i] = (__bf16)in[i];
}

// ---------------------------------------------------------------------------
// LayerNorm over channels (per pixel), write pixel-major bf16 activation
// x: [B, 256, HW] f32.  Aout[p*256 + c] bf16.  One wave per pixel.
// ---------------------------------------------------------------------------
__global__ void k_ln2d(const float* __restrict__ x, const float* __restrict__ w,
                       const float* __restrict__ bias, __bf16* __restrict__ Aout) {
    const int lane = threadIdx.x & 31;
    const int wave = threadIdx.x >> 5;
    const int p = blockIdx.x * 8 + wave;            // 8192 pixels
    const int b = p >> 10, hw = p & 1023;
    const float* xp = x + (size_t)b * C_ * HW_ + hw;
    float v[8], s = 0.f, ss = 0.f;
#pragma unroll
    for (int i = 0; i < 8; ++i) {
        v[i] = xp[(size_t)(lane + 32 * i) * HW_];
        s += v[i]; ss += v[i] * v[i];
    }
#pragma unroll
    for (int m = 16; m >= 1; m >>= 1) {
        s  += __shfl_xor(s,  m, 32);
        ss += __shfl_xor(ss, m, 32);
    }
    const float mean = s * (1.0f / C_);
    const float var  = ss * (1.0f / C_) - mean * mean;
    const float rstd = rsqrtf(var + 1e-6f);
#pragma unroll
    for (int i = 0; i < 8; ++i) {
        int c = lane + 32 * i;
        Aout[(size_t)p * C_ + c] = (__bf16)((v[i] - mean) * rstd * w[c] + bias[c]);
    }
}

// ---------------------------------------------------------------------------
// GEMM: out[b, o, hw] = sum_c A[p, c] * W[o, c] + bias[o] (+ resid)
// A: [8192, 256] bf16 row-major, W: [O, 256] bf16 row-major.
// One wave computes a 2x2 block of 16x16 tiles (32 pixels x 32 channels):
// each K step loads 2 A-frags + 2 B-frags and issues 4 WMMAs (2x frag reuse).
// ---------------------------------------------------------------------------
__global__ void k_gemm_bf16(const __bf16* __restrict__ A, const __bf16* __restrict__ W,
                            const float* __restrict__ bias, const float* __restrict__ resid,
                            float* __restrict__ out, int O) {
    const int lane = threadIdx.x & 31;
    const int wave = threadIdx.x >> 5;
    const int wid  = blockIdx.x * 8 + wave;
    const int tilesP2 = NPIX / 32;                  // 256 (pairs of pixel tiles)
    if (wid >= tilesP2 * (O >> 5)) return;
    const int tp2 = wid % tilesP2;
    const int to2 = wid / tilesP2;
    const int half = lane >> 4;
    const int l16  = lane & 15;

    const int rowA0 = tp2 * 32 + l16;               // pixel rows of the two A tiles
    const int rowA1 = rowA0 + 16;
    const int oc0   = to2 * 32 + l16;               // output channels of the two B tiles
    const int oc1   = oc0 + 16;

    const __bf16* Ap0 = A + (size_t)rowA0 * C_ + half * 16;
    const __bf16* Ap1 = A + (size_t)rowA1 * C_ + half * 16;
    const __bf16* Wp0 = W + (size_t)oc0 * C_ + half * 16;
    const __bf16* Wp1 = W + (size_t)oc1 * C_ + half * 16;

    v8f acc00 = {}, acc01 = {}, acc10 = {}, acc11 = {};
    for (int kc = 0; kc < C_; kc += 32) {
        v16bf a0, a1, b0, b1;
#pragma unroll
        for (int j = 0; j < 8; ++j) {
            a0[j]     = Ap0[kc + j];
            a0[8 + j] = Ap0[kc + 8 + j];
            a1[j]     = Ap1[kc + j];
            a1[8 + j] = Ap1[kc + 8 + j];
        }
#pragma unroll
        for (int j = 0; j < 16; ++j) {
            b0[j] = Wp0[kc + j];
            b1[j] = Wp1[kc + j];
        }
        acc00 = __builtin_amdgcn_wmma_f32_16x16x32_bf16(false, a0, false, b0, (short)0, acc00, false, false);
        acc01 = __builtin_amdgcn_wmma_f32_16x16x32_bf16(false, a0, false, b1, (short)0, acc01, false, false);
        acc10 = __builtin_amdgcn_wmma_f32_16x16x32_bf16(false, a1, false, b0, (short)0, acc10, false, false);
        acc11 = __builtin_amdgcn_wmma_f32_16x16x32_bf16(false, a1, false, b1, (short)0, acc11, false, false);
    }

    const float bv0 = bias[oc0];
    const float bv1 = bias[oc1];
#pragma unroll
    for (int ti = 0; ti < 2; ++ti) {                // pixel tile 0 / 1
        const int p0  = tp2 * 32 + ti * 16 + half * 8;
        const int bb  = p0 >> 10;
        const int hw0 = p0 & 1023;
        const size_t ob0 = ((size_t)bb * O + oc0) * HW_ + hw0;
        const size_t ob1 = ((size_t)bb * O + oc1) * HW_ + hw0;
        const v8f acA = ti ? acc10 : acc00;
        const v8f acB = ti ? acc11 : acc01;
#pragma unroll
        for (int r = 0; r < 8; ++r) {
            float v0 = acA[r] + bv0;
            float v1 = acB[r] + bv1;
            if (resid) { v0 += resid[ob0 + r]; v1 += resid[ob1 + r]; }
            out[ob0 + r] = v0;
            out[ob1 + r] = v1;
        }
    }
}

// ---------------------------------------------------------------------------
// Multi-scale depthwise conv + residual.
// in: [B, Cstride, HW] (channels coff .. coff+C-1), out: [B, C, HW]
// 4 channel groups of C/4 with kernel sizes 1,3,5,7.
// ---------------------------------------------------------------------------
__global__ void k_msdw(const float* __restrict__ in, float* __restrict__ out,
                       int C, int Cstride, int coff,
                       const float* __restrict__ w0, const float* __restrict__ w1,
                       const float* __restrict__ w2, const float* __restrict__ w3,
                       const float* __restrict__ b0, const float* __restrict__ b1,
                       const float* __restrict__ b2, const float* __restrict__ b3) {
    const int idx = blockIdx.x * 256 + threadIdx.x; // B*C*HW threads
    const int hw = idx & 1023;
    const int c  = (idx >> 10) % C;
    const int b  = idx / (C * HW_);
    const int h = hw >> 5, wc = hw & 31;
    const int g = C >> 2;
    const int grp = c / g, cl = c % g;
    const float* wp; const float* bp; int k;
    if      (grp == 0) { wp = w0; bp = b0; k = 1; }
    else if (grp == 1) { wp = w1; bp = b1; k = 3; }
    else if (grp == 2) { wp = w2; bp = b2; k = 5; }
    else               { wp = w3; bp = b3; k = 7; }
    const int pad = k >> 1;
    const float* xin = in + ((size_t)b * Cstride + coff + c) * HW_;
    float acc = bp[cl];
    for (int dy = -pad; dy <= pad; ++dy) {
        int hy = h + dy; if (hy < 0 || hy >= 32) continue;
        for (int dx = -pad; dx <= pad; ++dx) {
            int wx = wc + dx; if (wx < 0 || wx >= 32) continue;
            acc += xin[hy * 32 + wx] * wp[cl * k * k + (dy + pad) * k + (dx + pad)];
        }
    }
    out[((size_t)b * C + c) * HW_ + hw] = xin[hw] + acc;
}

// ---------------------------------------------------------------------------
// Flash attention. Q,K: [B,256,HW] f32 (dwconv'd). V lives inside QKV buffer
// at channel offset 512 with stride 768. One wave per 16-query tile per
// (b, head). Emits softmax(QK^T/sqrt(hd)) @ V directly as bf16 pixel-major A.
// ---------------------------------------------------------------------------
__global__ void k_attn(const float* __restrict__ Q, const float* __restrict__ K,
                       const float* __restrict__ V, __bf16* __restrict__ Aout) {
    __shared__ float lds[4][16 * 32];
    const int lane = threadIdx.x & 31;
    const int wave = threadIdx.x >> 5;
    const int wid = blockIdx.x * 4 + wave;          // 4096 waves
    const int tq = wid & 63;
    const int n  = (wid >> 6) & 7;
    const int b  = wid >> 9;
    const int half = lane >> 4;
    const int l16  = lane & 15;
    const float scale = 0.17677669529663687f;       // 1/sqrt(32)

    // Q fragment: A[t, d] 16x32
    v16bf aq;
    {
        const int t = tq * 16 + l16;
        const float* qp = Q + ((size_t)b * C_ + n * HD_) * HW_ + t;
        const int kb = half * 16;
#pragma unroll
        for (int j = 0; j < 8; ++j) aq[j]     = (__bf16)(qp[(size_t)(kb + j) * HW_] * scale);
#pragma unroll
        for (int j = 0; j < 8; ++j) aq[8 + j] = (__bf16)(qp[(size_t)(kb + 8 + j) * HW_] * scale);
    }

    float m[8], l[8];
    v8f acc0 = {}, acc1 = {};
#pragma unroll
    for (int r = 0; r < 8; ++r) { m[r] = -INFINITY; l[r] = 0.f; }

    const float* kbase = K + ((size_t)b * C_ + n * HD_) * HW_;
    const float* vbase = V + ((size_t)b * 768 + 512 + n * HD_) * HW_;
    float* pl = lds[wave];

    for (int ks = 0; ks < HW_; ks += 32) {
        // K fragments: B[d][s], two 16-key tiles
        v16bf bk0, bk1;
#pragma unroll
        for (int j = 0; j < 16; ++j) bk0[j] = (__bf16)kbase[(size_t)(half * 16 + j) * HW_ + ks + l16];
#pragma unroll
        for (int j = 0; j < 16; ++j) bk1[j] = (__bf16)kbase[(size_t)(half * 16 + j) * HW_ + ks + 16 + l16];
        v8f s0 = {}, s1 = {};
        s0 = __builtin_amdgcn_wmma_f32_16x16x32_bf16(false, aq, false, bk0, (short)0, s0, false, false);
        s1 = __builtin_amdgcn_wmma_f32_16x16x32_bf16(false, aq, false, bk1, (short)0, s1, false, false);

        // online softmax update (rows striped: lane group of 16 holds 8 rows)
        float p0[8], p1[8];
#pragma unroll
        for (int r = 0; r < 8; ++r) {
            float mx = fmaxf(s0[r], s1[r]);
#pragma unroll
            for (int msk = 1; msk <= 8; msk <<= 1) mx = fmaxf(mx, __shfl_xor(mx, msk, 32));
            const float mnew = fmaxf(m[r], mx);
            const float sc = __expf(m[r] - mnew);
            p0[r] = __expf(s0[r] - mnew);
            p1[r] = __expf(s1[r] - mnew);
            float rs = p0[r] + p1[r];
#pragma unroll
            for (int msk = 1; msk <= 8; msk <<= 1) rs += __shfl_xor(rs, msk, 32);
            l[r] = l[r] * sc + rs;
            m[r] = mnew;
            acc0[r] *= sc;
            acc1[r] *= sc;
        }

        // stage P (C layout) -> LDS -> reload in A layout
#pragma unroll
        for (int r = 0; r < 8; ++r) {
            const int t = r + 8 * half;
            pl[t * 32 + l16]      = p0[r];
            pl[t * 32 + 16 + l16] = p1[r];
        }
        __builtin_amdgcn_wave_barrier();
        v16bf ap;
        {
            const int t = l16, kb = half * 16;
#pragma unroll
            for (int j = 0; j < 8; ++j) ap[j]     = (__bf16)pl[t * 32 + kb + j];
#pragma unroll
            for (int j = 0; j < 8; ++j) ap[8 + j] = (__bf16)pl[t * 32 + kb + 8 + j];
        }
        __builtin_amdgcn_wave_barrier();

        // V fragments: B[s][d], d split 0..15 / 16..31
        v16bf bv0, bv1;
#pragma unroll
        for (int j = 0; j < 16; ++j) bv0[j] = (__bf16)vbase[(size_t)l16 * HW_ + ks + half * 16 + j];
#pragma unroll
        for (int j = 0; j < 16; ++j) bv1[j] = (__bf16)vbase[(size_t)(16 + l16) * HW_ + ks + half * 16 + j];
        acc0 = __builtin_amdgcn_wmma_f32_16x16x32_bf16(false, ap, false, bv0, (short)0, acc0, false, false);
        acc1 = __builtin_amdgcn_wmma_f32_16x16x32_bf16(false, ap, false, bv1, (short)0, acc1, false, false);
    }

    // epilogue: divide by row sum, write bf16 pixel-major A for proj GEMM
#pragma unroll
    for (int r = 0; r < 8; ++r) {
        const float inv = 1.0f / l[r];
        const int t = tq * 16 + r + 8 * half;
        const size_t p = (size_t)b * HW_ + t;
        Aout[p * C_ + n * HD_ + l16]      = (__bf16)(acc0[r] * inv);
        Aout[p * C_ + n * HD_ + 16 + l16] = (__bf16)(acc1[r] * inv);
    }
}

// ---------------------------------------------------------------------------
// Star gate: gelu(h1)*h2, h: [B,512,HW] -> bf16 pixel-major A[p,256]
// ---------------------------------------------------------------------------
__global__ void k_glu(const float* __restrict__ h, __bf16* __restrict__ Aout) {
    const int idx = blockIdx.x * 256 + threadIdx.x; // B*256*HW
    const int hw = idx & 1023;
    const int c  = (idx >> 10) & 255;
    const int b  = idx >> 18;
    const float x1 = h[((size_t)b * HID_ + c) * HW_ + hw];
    const float x2 = h[((size_t)b * HID_ + 256 + c) * HW_ + hw];
    const float g = 0.5f * x1 * (1.0f + erff(x1 * 0.70710678118654752f));
    const size_t p = (size_t)b * HW_ + hw;
    Aout[p * C_ + c] = (__bf16)(g * x2);
}

// ---------------------------------------------------------------------------
extern "C" void kernel_launch(void* const* d_in, const int* in_sizes, int n_in,
                              void* d_out, int out_size, void* d_ws, size_t ws_size,
                              hipStream_t stream) {
    (void)in_sizes; (void)n_in; (void)out_size; (void)ws_size;
    const float* x      = (const float*)d_in[0];
    const float* ln1_w  = (const float*)d_in[1];
    const float* ln1_b  = (const float*)d_in[2];
    const float* qkv_w  = (const float*)d_in[3];
    const float* qkv_b  = (const float*)d_in[4];
    const float* q_ws[4] = {(const float*)d_in[5],  (const float*)d_in[6],
                            (const float*)d_in[7],  (const float*)d_in[8]};
    const float* q_bs[4] = {(const float*)d_in[9],  (const float*)d_in[10],
                            (const float*)d_in[11], (const float*)d_in[12]};
    const float* k_ws[4] = {(const float*)d_in[13], (const float*)d_in[14],
                            (const float*)d_in[15], (const float*)d_in[16]};
    const float* k_bs[4] = {(const float*)d_in[17], (const float*)d_in[18],
                            (const float*)d_in[19], (const float*)d_in[20]};
    const float* proj_w = (const float*)d_in[21];
    const float* proj_b = (const float*)d_in[22];
    const float* ln2_w  = (const float*)d_in[23];
    const float* ln2_b  = (const float*)d_in[24];
    const float* fc1_w  = (const float*)d_in[25];
    const float* fc1_b  = (const float*)d_in[26];
    const float* m_ws[4] = {(const float*)d_in[27], (const float*)d_in[28],
                            (const float*)d_in[29], (const float*)d_in[30]};
    const float* m_bs[4] = {(const float*)d_in[31], (const float*)d_in[32],
                            (const float*)d_in[33], (const float*)d_in[34]};
    const float* fc2_w  = (const float*)d_in[35];
    const float* fc2_b  = (const float*)d_in[36];
    float* out = (float*)d_out;

    char* ws = (char*)d_ws;
    __bf16* wqkv  = (__bf16*)(ws);                    // 768*256*2  = 393216
    __bf16* wproj = (__bf16*)(ws + 393216);           // 256*256*2  = 131072
    __bf16* wfc1  = (__bf16*)(ws + 524288);           // 512*256*2  = 262144
    __bf16* wfc2  = (__bf16*)(ws + 786432);           // 256*256*2  = 131072
    __bf16* Abf   = (__bf16*)(ws + 1048576);          // 8192*256*2 = 4 MB
    float*  QKV   = (float*)(ws + 5242880);           // 8*768*1024*4 = 24 MB
    float*  QP    = (float*)(ws + 30408704);          // 8 MB
    float*  KP    = (float*)(ws + 38797312);          // 8 MB
    float*  X2    = (float*)(ws + 47185920);          // 8 MB
    float*  HMLP  = (float*)(ws + 55574528);          // 16 MB
    float*  HMLP2 = (float*)(ws + 72351744);          // 16 MB

    // 1) cast weights to bf16
    k_cast_bf16<<<(768 * 256 + 255) / 256, 256, 0, stream>>>(qkv_w, wqkv, 768 * 256);
    k_cast_bf16<<<(256 * 256 + 255) / 256, 256, 0, stream>>>(proj_w, wproj, 256 * 256);
    k_cast_bf16<<<(512 * 256 + 255) / 256, 256, 0, stream>>>(fc1_w, wfc1, 512 * 256);
    k_cast_bf16<<<(256 * 256 + 255) / 256, 256, 0, stream>>>(fc2_w, wfc2, 256 * 256);

    // 2) LN1 -> Abf, QKV GEMM (O=768): 256 * (768/32) waves / 8 per block
    k_ln2d<<<NPIX / 8, 256, 0, stream>>>(x, ln1_w, ln1_b, Abf);
    k_gemm_bf16<<<(NPIX / 32) * (768 / 32) / 8, 256, 0, stream>>>(Abf, wqkv, qkv_b, nullptr, QKV, 768);

    // 3) q/k multi-scale depthwise + residual
    k_msdw<<<B_ * C_ * HW_ / 256, 256, 0, stream>>>(QKV, QP, C_, 768, 0,
        q_ws[0], q_ws[1], q_ws[2], q_ws[3], q_bs[0], q_bs[1], q_bs[2], q_bs[3]);
    k_msdw<<<B_ * C_ * HW_ / 256, 256, 0, stream>>>(QKV, KP, C_, 768, 256,
        k_ws[0], k_ws[1], k_ws[2], k_ws[3], k_bs[0], k_bs[1], k_bs[2], k_bs[3]);

    // 4) flash attention -> bf16 A; proj GEMM with x residual -> X2
    k_attn<<<B_ * NH_ * (HW_ / 16) / 4, 128, 0, stream>>>(QP, KP, QKV, Abf);
    k_gemm_bf16<<<(NPIX / 32) * (256 / 32) / 8, 256, 0, stream>>>(Abf, wproj, proj_b, x, X2, 256);

    // 5) FFN: LN2, fc1 (O=512), msdw, gelu-gate, fc2 (+X2 residual) -> d_out
    k_ln2d<<<NPIX / 8, 256, 0, stream>>>(X2, ln2_w, ln2_b, Abf);
    k_gemm_bf16<<<(NPIX / 32) * (512 / 32) / 8, 256, 0, stream>>>(Abf, wfc1, fc1_b, nullptr, HMLP, 512);
    k_msdw<<<B_ * HID_ * HW_ / 256, 256, 0, stream>>>(HMLP, HMLP2, HID_, HID_, 0,
        m_ws[0], m_ws[1], m_ws[2], m_ws[3], m_bs[0], m_bs[1], m_bs[2], m_bs[3]);
    k_glu<<<B_ * C_ * HW_ / 256, 256, 0, stream>>>(HMLP2, Abf);
    k_gemm_bf16<<<(NPIX / 32) * (256 / 32) / 8, 256, 0, stream>>>(Abf, wfc2, fc2_b, X2, out, 256);
}
